// Attention_69518340653440
// MI455X (gfx1250) — compile-verified, matrix operand order checked by reading, not run on hardware
//
#include <hip/hip_runtime.h>
#include <hip/hip_bf16.h>

#define DIM    1024
#define NHEAD  16
#define HDIM   64
#define BATCH  2
#define SEQ    2048
#define TOKENS (BATCH * SEQ)   // 4096
#define LN_EPS 1e-5f

typedef __attribute__((ext_vector_type(16))) _Float16 v16h;
typedef __attribute__((ext_vector_type(8)))  _Float16 v8h;
typedef __attribute__((ext_vector_type(4)))  _Float16 v4h;
typedef __attribute__((ext_vector_type(8)))  float    v8f;
typedef __attribute__((ext_vector_type(4)))  float    v4f;

// D = A(16x32 f16) x B(32x16 f16) + C(16x16 f32), wave32 WMMA
#define WMMA_F16(A, B, C) \
  __builtin_amdgcn_wmma_f32_16x16x32_f16(false, (A), false, (B), (short)0, (C), false, false)

// ---- fragment loader --------------------------------------------------------
// 16-bit A/B fragment element pattern (per CDNA5 ISA):
//   lane L, khb = (L>>4)*8 ; element e<8 -> K = khb+e ; e>=8 -> K = 16+khb+(e-8)
static __device__ inline v16h frag_f16(const _Float16* __restrict__ row, int khb) {
  v8h lo = *(const v8h*)(row + khb);
  v8h hi = *(const v8h*)(row + 16 + khb);
  v16h f;
#pragma unroll
  for (int i = 0; i < 8; ++i) { f[i] = lo[i]; f[i + 8] = hi[i]; }
  return f;
}

// ---- Kernel 0: one-shot f32 -> f16 pack (exact grid, no tail) ---------------
__global__ void __launch_bounds__(256) pack_f16(
    const float* __restrict__ src, _Float16* __restrict__ dst) {
  const int i = blockIdx.x * 256 + threadIdx.x;   // float4 index
  v4f a = ((const v4f*)src)[i];
  v4h r;
#pragma unroll
  for (int j = 0; j < 4; ++j) r[j] = (_Float16)a[j];
  ((v4h*)dst)[i] = r;
}

// ---- Kernel 1: C = A @ W^T + bias  (f16 A, f16 W, f32 out) ------------------
// Wave tile 32(M) x 64(N): 2 A-frags + 4 B-frags -> 8 WMMA per K-slice.
// Ping-pong double buffer (manual 2x unroll): no register-rotation moves,
// next slice's 12 b128 loads always in flight behind 8 WMMAs.
__global__ void __launch_bounds__(256) gemm_f16(
    const _Float16* __restrict__ A, const _Float16* __restrict__ W,
    const float* __restrict__ bias, float* __restrict__ C,
    int N, int K) {
  const int lane = threadIdx.x & 31;
  const int wave = threadIdx.x >> 5;
  const int l16  = lane & 15;
  const int khb  = (lane >> 4) * 8;
  // block = 8 waves: 4 in M x 2 in N = 128(M) x 128(N)
  const int rowTile = blockIdx.x * 128 + (wave >> 1) * 32;
  const int colTile = blockIdx.y * 128 + (wave & 1) * 64;

  const _Float16* Arow0 = A + (size_t)(rowTile + l16) * K;
  const _Float16* Arow1 = Arow0 + (size_t)16 * K;
  const _Float16* Wrow[4];
#pragma unroll
  for (int nt = 0; nt < 4; ++nt)
    Wrow[nt] = W + (size_t)(colTile + nt * 16 + l16) * K;

  v8f acc[2][4];
#pragma unroll
  for (int nt = 0; nt < 4; ++nt) {
    float bv = bias[colTile + nt * 16 + l16];   // C-tile column = l16
#pragma unroll
    for (int i = 0; i < 8; ++i) { acc[0][nt][i] = bv; acc[1][nt][i] = bv; }
  }

  v16h a0A, a1A, bA[4];   // buffer A
  v16h a0B, a1B, bB[4];   // buffer B

  auto load_slice = [&](int kk, v16h& A0, v16h& A1, v16h* B) {
    A0 = frag_f16(Arow0 + kk, khb);
    A1 = frag_f16(Arow1 + kk, khb);
#pragma unroll
    for (int nt = 0; nt < 4; ++nt) B[nt] = frag_f16(Wrow[nt] + kk, khb);
  };
  auto do_wmma = [&](v16h& A0, v16h& A1, v16h* B) {
#pragma unroll
    for (int nt = 0; nt < 4; ++nt) {
      acc[0][nt] = WMMA_F16(A0, B[nt], acc[0][nt]);
      acc[1][nt] = WMMA_F16(A1, B[nt], acc[1][nt]);
    }
  };

  load_slice(0, a0A, a1A, bA);
  int kk = 32;
  for (; kk < K - 32; kk += 64) {           // K % 64 == 0, K >= 64
    load_slice(kk, a0B, a1B, bB);
    __builtin_prefetch(Arow0 + kk + 64, 0, 1);   // global_prefetch_b8
    do_wmma(a0A, a1A, bA);
    load_slice(kk + 32, a0A, a1A, bA);
    do_wmma(a0B, a1B, bB);
  }
  load_slice(kk, a0B, a1B, bB);             // kk == K-32 here
  do_wmma(a0A, a1A, bA);
  do_wmma(a0B, a1B, bB);

#pragma unroll
  for (int mr = 0; mr < 2; ++mr)
#pragma unroll
    for (int nt = 0; nt < 4; ++nt)
#pragma unroll
      for (int r = 0; r < 8; ++r) {
        int row = rowTile + mr * 16 + r + khb;    // M = r + 8*(lane>>4)
        int col = colTile + nt * 16 + l16;
        C[(size_t)row * N + col] = acc[mr][nt][r];
      }
}

// ---- Kernel 2: per-head qk LayerNorm -> f16 [B,H,N,D] -----------------------
// One wave per (q|k, b, h, n) row; D=64 -> 2 elems/lane, shfl_xor reductions.
__global__ void __launch_bounds__(256) ln_qk(
    const float* __restrict__ qkv,
    const float* __restrict__ qg, const float* __restrict__ qb,
    const float* __restrict__ kg, const float* __restrict__ kb,
    _Float16* __restrict__ qh, _Float16* __restrict__ kh) {
  const int lane = threadIdx.x & 31;
  const int w    = blockIdx.x * 8 + (threadIdx.x >> 5);
  const int sec  = w >> 16;            // 0=q, 1=k (65536 rows each)
  const int r    = w & 65535;
  const int b    = r >> 15;
  const int h    = (r >> 11) & 15;
  const int n    = r & 2047;
  const int d0   = lane * 2;

  const float* src = qkv + (size_t)(b * SEQ + n) * (3 * DIM) + sec * DIM + h * HDIM;
  float x0 = src[d0], x1 = src[d0 + 1];

  float s = x0 + x1;
#pragma unroll
  for (int m = 16; m >= 1; m >>= 1) s += __shfl_xor(s, m, 32);
  float mu  = s * (1.0f / 64.0f);
  float dx0 = x0 - mu, dx1 = x1 - mu;
  float sq = dx0 * dx0 + dx1 * dx1;
#pragma unroll
  for (int m = 16; m >= 1; m >>= 1) sq += __shfl_xor(sq, m, 32);
  float rstd = rsqrtf(sq * (1.0f / 64.0f) + LN_EPS);

  const float* g  = (sec == 0) ? qg : kg;
  const float* be = (sec == 0) ? qb : kb;
  float y0 = dx0 * rstd * g[d0]     + be[d0];
  float y1 = dx1 * rstd * g[d0 + 1] + be[d0 + 1];
  if (sec == 0) { y0 *= 0.125f; y1 *= 0.125f; }   // fold attn scale 1/sqrt(64) into q

  _Float16* dst = ((sec == 0) ? qh : kh) + (size_t)((b * NHEAD + h) * SEQ + n) * HDIM;
  dst[d0]     = (_Float16)y0;
  dst[d0 + 1] = (_Float16)y1;
}

// ---- Kernel 3: v transpose via LDS -> f16 vT[B,H,D,N] -----------------------
// Block: one (b,h) x 64-token slab. 64x64 f32 tile in LDS (stride 65 ->
// conflict-free column reads), 32B vectorized f16 stores along n.
__global__ void __launch_bounds__(256) v_transpose(
    const float* __restrict__ qkv, _Float16* __restrict__ vT) {
  __shared__ float tile[64][65];
  const int bh   = blockIdx.x >> 5;    // 0..31
  const int slab = blockIdx.x & 31;    // n-slab of 64
  const int b    = bh >> 4;
  const int h    = bh & 15;
  const int t    = threadIdx.x;

  // load 64 rows (n) x 64 (d): 4 threads per row, 16 floats each
  {
    const int row = t >> 2;
    const int dq  = (t & 3) * 16;
    const int n   = slab * 64 + row;
    const float* src = qkv + (size_t)(b * SEQ + n) * (3 * DIM) + 2 * DIM + h * HDIM + dq;
#pragma unroll
    for (int j = 0; j < 4; ++j) {
      v4f a = *(const v4f*)(src + j * 4);
#pragma unroll
      for (int c = 0; c < 4; ++c) tile[row][dq + j * 4 + c] = a[c];
    }
  }
  __syncthreads();
  // store: vT[bh][d][slab*64 + nq .. +15], 16 halves = 32B per thread
  {
    const int d  = t >> 2;
    const int nq = (t & 3) * 16;
    v16h r;
#pragma unroll
    for (int j = 0; j < 16; ++j) r[j] = (_Float16)tile[nq + j][d];
    _Float16* dst = vT + (size_t)bh * HDIM * SEQ + (size_t)d * SEQ + slab * 64 + nq;
    *(v16h*)dst = r;
  }
}

// ---- Kernel 4: flash attention, transposed tiles ----------------------------
// Per wave: 16 queries x all 2048 keys. Computes S^T = K x Q^T so that
//   (a) softmax reductions are per-column (= per lane&15): 8 VALU + 1 shfl_xor(16)
//   (b) exp'd S^T C-tiles map register-for-register onto the B-fragment of
//       O^T = vT x P^T (element e<8 <- tile0[e], e>=8 <- tile1[e-8], same lane).
// All k/v loads for a chunk issue up front; score WMMAs use partial loadcnt
// waits while v loads stay in flight behind the softmax VALU chain.
__global__ void __launch_bounds__(256) attn(
    const _Float16* __restrict__ qh, const _Float16* __restrict__ kh,
    const _Float16* __restrict__ vT, _Float16* __restrict__ ao) {
  const int lane = threadIdx.x & 31;
  const int w    = blockIdx.x * 8 + (threadIdx.x >> 5);   // 0..4095
  const int bh   = w >> 7;                                // 0..31
  const int qt   = w & 127;
  const int b    = bh >> 4, h = bh & 15;
  const int l16  = lane & 15;
  const int khb  = (lane >> 4) * 8;

  const _Float16* qrow = qh + (size_t)(bh * SEQ + qt * 16 + l16) * HDIM;
  v16h qlo = frag_f16(qrow, khb);        // d = 0..31
  v16h qhi = frag_f16(qrow + 32, khb);   // d = 32..63

  float m = -3.0e38f, l = 0.0f;
  v8f acc[4];
#pragma unroll
  for (int mt = 0; mt < 4; ++mt) {
    v8f z = {};
    acc[mt] = z;
  }

  const _Float16* kbase = kh + (size_t)bh * SEQ * HDIM;
  const _Float16* vbase = vT + (size_t)bh * HDIM * SEQ;

  for (int kc = 0; kc < SEQ; kc += 32) {
    const _Float16* kr0 = kbase + (size_t)(kc + l16) * HDIM;
    const _Float16* kr1 = kr0 + 16 * HDIM;
    // K-tile fragments (issued first)
    v16h a0lo = frag_f16(kr0, khb), a0hi = frag_f16(kr0 + 32, khb);
    v16h a1lo = frag_f16(kr1, khb), a1hi = frag_f16(kr1 + 32, khb);
    // V-tile fragments hoisted: in flight during score WMMAs + softmax
    v16h va[4];
#pragma unroll
    for (int mt = 0; mt < 4; ++mt)
      va[mt] = frag_f16(vbase + (size_t)(mt * 16 + l16) * SEQ + kc, khb);
    __builtin_prefetch(kr0 + 32 * HDIM, 0, 1);

    v8f z = {};
    v8f st0 = WMMA_F16(a0lo, qlo, z);  st0 = WMMA_F16(a0hi, qhi, st0);
    v8f st1 = WMMA_F16(a1lo, qlo, z);  st1 = WMMA_F16(a1hi, qhi, st1);

    // per-query (column) max
    float tm = st0[0];
#pragma unroll
    for (int i = 0; i < 8; ++i) { tm = fmaxf(tm, st0[i]); tm = fmaxf(tm, st1[i]); }
    tm = fmaxf(tm, __shfl_xor(tm, 16, 32));
    float mn   = fmaxf(m, tm);
    float corr = __expf(m - mn);
    m = mn;

    float rs = 0.0f;
#pragma unroll
    for (int i = 0; i < 8; ++i) {
      st0[i] = __expf(st0[i] - mn);
      st1[i] = __expf(st1[i] - mn);
      rs += st0[i] + st1[i];
    }
    rs += __shfl_xor(rs, 16, 32);
    l = l * corr + rs;

    // P^T tiles -> B-fragment, pure in-lane repack
    v16h pb;
#pragma unroll
    for (int i = 0; i < 8; ++i) { pb[i] = (_Float16)st0[i]; pb[i + 8] = (_Float16)st1[i]; }

#pragma unroll
    for (int mt = 0; mt < 4; ++mt) {
#pragma unroll
      for (int i = 0; i < 8; ++i) acc[mt][i] *= corr;
      acc[mt] = WMMA_F16(va[mt], pb, acc[mt]);   // O^T[d,q] += vT[d,k] * P^T[k,q]
    }
  }

  const float invl = 1.0f / l;
  const int n = qt * 16 + l16;                               // query index
  _Float16* orow = ao + (size_t)(b * SEQ + n) * DIM + h * HDIM;
#pragma unroll
  for (int mt = 0; mt < 4; ++mt)
#pragma unroll
    for (int r = 0; r < 8; ++r) {
      int d = mt * 16 + r + khb;
      orow[d] = (_Float16)(acc[mt][r] * invl);
    }
}

// ---- launch -----------------------------------------------------------------
extern "C" void kernel_launch(void* const* d_in, const int* in_sizes, int n_in,
                              void* d_out, int out_size, void* d_ws, size_t ws_size,
                              hipStream_t stream) {
  const float* x       = (const float*)d_in[0];
  const float* w_qkv   = (const float*)d_in[1];
  const float* b_qkv   = (const float*)d_in[2];
  const float* q_gamma = (const float*)d_in[3];
  const float* q_beta  = (const float*)d_in[4];
  const float* k_gamma = (const float*)d_in[5];
  const float* k_beta  = (const float*)d_in[6];
  const float* w_proj  = (const float*)d_in[7];
  const float* b_proj  = (const float*)d_in[8];
  float* out = (float*)d_out;

  // workspace layout (96 MiB total, all offsets 256B-aligned)
  char* ws = (char*)d_ws;
  const size_t QKV_BYTES   = (size_t)TOKENS * 3 * DIM * sizeof(float);   // 48 MiB
  const size_t XH_BYTES    = (size_t)TOKENS * DIM * 2;                   // 8 MiB
  const size_t WQKV_BYTES  = (size_t)3 * DIM * DIM * 2;                  // 6 MiB
  const size_t WPROJ_BYTES = (size_t)DIM * DIM * 2;                      // 2 MiB
  const size_t H16_BYTES   = (size_t)BATCH * NHEAD * SEQ * HDIM * 2;     // 8 MiB each

  size_t off = 0;
  float*    qkv    = (float*)(ws + off);    off += QKV_BYTES;
  _Float16* xh     = (_Float16*)(ws + off); off += XH_BYTES;
  _Float16* wqkvh  = (_Float16*)(ws + off); off += WQKV_BYTES;
  _Float16* wprojh = (_Float16*)(ws + off); off += WPROJ_BYTES;
  _Float16* qhd    = (_Float16*)(ws + off); off += H16_BYTES;
  _Float16* khd    = (_Float16*)(ws + off); off += H16_BYTES;
  _Float16* vT     = (_Float16*)(ws + off); off += H16_BYTES;
  _Float16* ao     = (_Float16*)(ws + off); off += H16_BYTES;

  dim3 blk(256);

  // one-shot f32->f16 packs (exact grids: counts are multiples of 1024 floats)
  pack_f16<<<dim3((TOKENS * DIM) / 1024), blk, 0, stream>>>(x, xh);
  pack_f16<<<dim3((3 * DIM * DIM) / 1024), blk, 0, stream>>>(w_qkv, wqkvh);
  pack_f16<<<dim3((DIM * DIM) / 1024), blk, 0, stream>>>(w_proj, wprojh);

  // qkv = x @ w_qkv^T + b_qkv   (wave tile 32x64, block tile 128x128)
  gemm_f16<<<dim3(TOKENS / 128, (3 * DIM) / 128), blk, 0, stream>>>(
      xh, wqkvh, b_qkv, qkv, 3 * DIM, DIM);

  ln_qk<<<dim3(2 * BATCH * NHEAD * SEQ / 8), blk, 0, stream>>>(
      qkv, q_gamma, q_beta, k_gamma, k_beta, qhd, khd);
  v_transpose<<<dim3(BATCH * NHEAD * (SEQ / 64)), blk, 0, stream>>>(qkv, vT);

  attn<<<dim3(BATCH * NHEAD * (SEQ / 16) / 8), blk, 0, stream>>>(qhd, khd, vT, ao);

  // out = ao @ w_proj^T + b_proj
  gemm_f16<<<dim3(TOKENS / 128, DIM / 128), blk, 0, stream>>>(
      ao, wprojh, b_proj, out, DIM, DIM);
}
